// GRUDINA_6296422056644
// MI455X (gfx1250) — matrix operand order repaired and checked by reading, not compiled, not run on hardware
//
#include <hip/hip_runtime.h>
#include <math.h>

typedef _Float16 half8 __attribute__((ext_vector_type(8)));
typedef _Float16 v16h  __attribute__((ext_vector_type(16)));
typedef float    v8f   __attribute__((ext_vector_type(8)));

#define BB   64
#define LL   200
#define DD   256
#define HH   512
#define QQ   1024
#define G3   1536
#define NQ_  1024
#define NROWS (BB*LL)      // 12800
#define HS16 520           // padded LDS stride (halves) to spread banks
#define HS32 516           // padded LDS stride (floats)

// ---------- WMMA fragment loaders (ISA 7.12.2 layouts, wave32) ----------
// A 16x32 f16: lane l holds row m=l&15; lanes<16: K {k..k+7, k+16..k+23};
// lanes>=16: K {k+8..k+15, k+24..k+31}. Two contiguous 16B loads.
static __device__ __forceinline__ v16h ldA(const _Float16* p) {
    half8 lo = *(const half8*)p;
    half8 hi = *(const half8*)(p + 16);
    return __builtin_shufflevector(lo, hi, 0,1,2,3,4,5,6,7,8,9,10,11,12,13,14,15);
}
// B 32x16 f16 with W stored N-major (N x K): lane l holds col n=l&15;
// lanes<16: K {k..k+7, k+16..k+23}; lanes>=16: K {k+8..k+15, k+24..k+31}. Two 16B loads.
static __device__ __forceinline__ v16h ldB(const _Float16* p) {
    half8 b0 = *(const half8*)p;
    half8 b1 = *(const half8*)(p + 16);
    return __builtin_shufflevector(b0, b1, 0,1,2,3,4,5,6,7,8,9,10,11,12,13,14,15);
}

static __device__ __forceinline__ float sigm(float x) {
    return 1.f / (1.f + __expf(-x));
}
static __device__ __forceinline__ float tanh_fast(float x) {
    float e = __expf(2.f * x);
    return 1.f - 2.f / (e + 1.f);
}

// ---------------- K0: convert weights to f16 (N x K natural layouts) ----
__global__ void conv_w_kernel(const float* wih, const float* whh, const float* fcw,
                              _Float16* wih16, _Float16* whh16, _Float16* fcw16) {
    int i = blockIdx.x * blockDim.x + threadIdx.x;
    if (i < G3 * HH) { wih16[i] = (_Float16)wih[i]; whh16[i] = (_Float16)whh[i]; }
    if (i < QQ * HH) { fcw16[i] = (_Float16)fcw[i]; }
}

// ---------------- K1: embeddings -> x (f16, [B*L][2D]) ------------------
__global__ void build_x_kernel(const int* q_data, const int* qa_data, const int* pid_data,
                               const float* q_emb, const float* qa_emb,
                               const float* q_emb_diff, const float* qa_emb_diff,
                               const float* diff_parm, _Float16* x16) {
    int i = blockIdx.x;                 // (b,l) flattened, 12800
    int d = threadIdx.x;                // 0..255
    int q   = q_data[i];
    int qa  = (qa_data[i] - q) / NQ_;   // 0/1
    float pe = diff_parm[pid_data[i]];
    float qe  = q_emb[(size_t)q * DD + d];
    float q2  = qe + pe * q_emb_diff[(size_t)q * DD + d];
    float qae = qa_emb[(size_t)qa * DD + d] + qe;
    float qa2 = qae + pe * qa_emb_diff[(size_t)qa * DD + d];
    x16[(size_t)i * (2*DD) + d]      = (_Float16)qa2;   // qa_embed2 first
    x16[(size_t)i * (2*DD) + DD + d] = (_Float16)q2;    // q_embed2 second
}

// ---------------- K2: xg = x @ w_ih^T + b_ih  ([L][B][3H] f32) ----------
__global__ void __launch_bounds__(256)
xg_gemm_kernel(const _Float16* x16, const _Float16* wih16, const float* bih, float* xg) {
    const int wave = threadIdx.x >> 5, lane = threadIdx.x & 31;
    const int l16 = lane >> 4, nl = lane & 15;
    const int mt = blockIdx.x;                       // 800 M-tiles
    const int nbase = blockIdx.y * 128 + wave * 16;  // N in [0,1536)
    v8f acc = {};
    const _Float16* arow = x16 + (size_t)(mt * 16 + nl) * (2*DD);
    const _Float16* brow = wih16 + (size_t)(nbase + nl) * (2*DD);
#pragma unroll
    for (int kk = 0; kk < 16; ++kk) {
        int k = kk * 32;
        v16h a = ldA(arow + k + (l16 ? 8 : 0));
        v16h b = ldB(brow + k + (l16 ? 8 : 0));
        acc = __builtin_amdgcn_wmma_f32_16x16x32_f16(false, a, false, b,
                                                     (short)0, acc, false, false);
    }
    int n = nbase + nl;
    float bias = bih[n];
#pragma unroll
    for (int e = 0; e < 8; ++e) {
        int r = mt * 16 + e + 8 * l16;               // global row b*L+t
        int bi = r / LL, ti = r - bi * LL;
        xg[((size_t)ti * BB + bi) * G3 + n] = acc[e] + bias;   // [L][B][3H]
    }
}

// ---------------- K3: GRU recurrence, 4 WGs x 16 batch rows -------------
__global__ void __launch_bounds__(512)
gru_kernel(const float* xg, const _Float16* whh16, const float* bhh, _Float16* gru16) {
    __shared__ _Float16 h16s[16 * HS16];
    __shared__ float    h32s[16 * HS32];
    const int tid = threadIdx.x;
    const int wave = tid >> 5, lane = tid & 31;
    const int l16 = lane >> 4, nl = lane & 15;
    const int chunk = blockIdx.x;                    // batch rows [chunk*16, +16)

    for (int i = tid; i < 16 * HS16; i += 512) h16s[i] = (_Float16)0.f;
    for (int i = tid; i < 16 * HS32; i += 512) h32s[i] = 0.f;
    __syncthreads();

    const int colbase = wave * 32;                   // H cols owned by this wave
    float bias[3][2];
#pragma unroll
    for (int g = 0; g < 3; ++g)
#pragma unroll
        for (int j = 0; j < 2; ++j)
            bias[g][j] = bhh[g * HH + colbase + j * 16 + nl];

    for (int t = 0; t < LL; ++t) {
        v8f acc[3][2];
#pragma unroll
        for (int g = 0; g < 3; ++g) { acc[g][0] = (v8f){}; acc[g][1] = (v8f){}; }

        for (int kk = 0; kk < 16; ++kk) {
            int k = kk * 32;
            // A frag from LDS-resident h (f16)
            int aoff = nl * HS16 + k + (l16 ? 8 : 0);
            half8 lo = *(half8*)&h16s[aoff];
            half8 hi = *(half8*)&h16s[aoff + 16];
            v16h a = __builtin_shufflevector(lo, hi, 0,1,2,3,4,5,6,7,8,9,10,11,12,13,14,15);
#pragma unroll
            for (int g = 0; g < 3; ++g)
#pragma unroll
                for (int j = 0; j < 2; ++j) {
                    int n = g * HH + colbase + j * 16 + nl;
                    v16h b = ldB(whh16 + (size_t)n * HH + k + (l16 ? 8 : 0));
                    acc[g][j] = __builtin_amdgcn_wmma_f32_16x16x32_f16(
                        false, a, false, b, (short)0, acc[g][j], false, false);
                }
        }

        // Gate math: this wave owns r/z/n for its own 32 H-columns -> no
        // cross-wave hg exchange needed.
        float hnew[2][8];
#pragma unroll
        for (int j = 0; j < 2; ++j) {
            int c = colbase + j * 16 + nl;
#pragma unroll
            for (int e = 0; e < 8; ++e) {
                int m = e + 8 * l16;
                size_t xrow = ((size_t)t * BB + chunk * 16 + m) * G3;
                float xr = xg[xrow + c];
                float xz = xg[xrow + HH + c];
                float xn = xg[xrow + 2 * HH + c];
                float hr = acc[0][j][e] + bias[0][j];
                float hz = acc[1][j][e] + bias[1][j];
                float hn = acc[2][j][e] + bias[2][j];
                float r = sigm(xr + hr);
                float z = sigm(xz + hz);
                float nn = tanh_fast(xn + r * hn);
                float hold = h32s[m * HS32 + c];
                hnew[j][e] = (1.f - z) * nn + z * hold;
            }
        }
        __syncthreads();   // everyone done reading old h
#pragma unroll
        for (int j = 0; j < 2; ++j) {
            int c = colbase + j * 16 + nl;
#pragma unroll
            for (int e = 0; e < 8; ++e) {
                int m = e + 8 * l16;
                float v = hnew[j][e];
                h32s[m * HS32 + c] = v;
                h16s[m * HS16 + c] = (_Float16)v;
                gru16[((size_t)(chunk * 16 + m) * LL + t) * HH + c] = (_Float16)v;
            }
        }
        __syncthreads();   // new h visible before next step's A frags
    }
}

// ---------------- K4: fc_out = gru @ fc_w^T + fc_b ([row][Q] f32) -------
__global__ void __launch_bounds__(256)
fc_gemm_kernel(const _Float16* gru16, const _Float16* fcw16, const float* fcb, float* fco) {
    const int wave = threadIdx.x >> 5, lane = threadIdx.x & 31;
    const int l16 = lane >> 4, nl = lane & 15;
    const int mt = blockIdx.x;
    const int nbase = blockIdx.y * 128 + wave * 16;  // N in [0,1024)
    v8f acc = {};
    const _Float16* arow = gru16 + (size_t)(mt * 16 + nl) * HH;
    const _Float16* brow = fcw16 + (size_t)(nbase + nl) * HH;
#pragma unroll
    for (int kk = 0; kk < 16; ++kk) {
        int k = kk * 32;
        v16h a = ldA(arow + k + (l16 ? 8 : 0));
        v16h b = ldB(brow + k + (l16 ? 8 : 0));
        acc = __builtin_amdgcn_wmma_f32_16x16x32_f16(false, a, false, b,
                                                     (short)0, acc, false, false);
    }
    int n = nbase + nl;
    float bias = fcb[n];
#pragma unroll
    for (int e = 0; e < 8; ++e) {
        int r = mt * 16 + e + 8 * l16;
        fco[(size_t)r * QQ + n] = acc[e] + bias;
    }
}

// ------- K5: mvals[b,t] = threshold( dot(fc_out[b,t,:], matrix[q-1,:]) ) -
// Only the asked question's mastery is consumed downstream, so the
// 26.8 GFLOP [B,L,Q]x[Q,Q] einsum collapses to 12800 length-1024 dots.
__global__ void __launch_bounds__(256)
mvals_kernel(const float* fco, const float* matrix, const int* q_data, float* mvals) {
    const int wave = threadIdx.x >> 5, lane = threadIdx.x & 31;
    const int row = blockIdx.x * 8 + wave;           // 1600 blocks
    const int idx = q_data[row] - 1;
    const float* a = fco + (size_t)row * QQ;
    const float* m = matrix + (size_t)idx * QQ;
    float s = 0.f;
    for (int i = lane; i < QQ; i += 32) s += a[i] * m[i];
#pragma unroll
    for (int off = 16; off; off >>= 1) s += __shfl_xor(s, off, 32);
    if (lane == 0) mvals[row] = (s >= 0.4f) ? 1.f : s;
}

// ---------------- K6: pairwise history stats per batch row --------------
__global__ void __launch_bounds__(256)
stats_kernel(const int* q_data, const int* qa_data, const float* mvals,
             float* mc, float* mi, float* nmc, float* aa) {
    const int b = blockIdx.x, tid = threadIdx.x;
    __shared__ int   qs[LL];
    __shared__ float qa1[LL], ma[LL], nm[LL];
    if (tid < LL) {
        int i = b * LL + tid;
        int q = q_data[i];
        qs[tid] = q;
        qa1[tid] = (float)((qa_data[i] - q) / NQ_);
        float m = mvals[i];
        ma[tid] = (m == 1.f) ? 1.f : 0.f;
        nm[tid] = (m == 0.f) ? 1.f : 0.f;
    }
    __syncthreads();
    if (tid < LL) {
        int qt = qs[tid];
        float fmc = 0.f, fmi = 0.f, fn = 0.f, fa = 0.f;
        for (int k = 0; k < tid; ++k)
            if (qs[k] == qt) {
                fmc += qa1[k] * ma[k];
                fmi += (1.f - qa1[k]) * ma[k];
                fn  += (1.f - qa1[k]) * nm[k];
                fa  += 1.f;
            }
        int i = b * LL + tid;
        mc[i] = fmc; mi[i] = fmi; nmc[i] = fn; aa[i] = fa + 1.f;  // k<=t incl self
    }
}

// ---------------- K7: DINA scan + sigmoid preds + loss (deterministic) --
__global__ void __launch_bounds__(64)
dina_kernel(const int* q_data, const int* qa_data, const int* pid_data,
            const float* diff_parm, const float* target, const float* mvals,
            const float* mc, const float* mi, const float* nmc, const float* aa,
            float* guess, float* slip, float* out) {
    const int b = threadIdx.x;   // one thread per batch row, 64 threads
    __shared__ float s_mse[BB], s_creg[BB];
    __shared__ int   s_cnt[BB];
    float* gRow = guess + (size_t)b * QQ;
    float* sRow = slip  + (size_t)b * QQ;
    for (int q = 0; q < QQ; ++q) { gRow[q] = 0.f; sRow[q] = 0.f; }
    float mse = 0.f, creg = 0.f; int cnt = 0;
    for (int t = 0; t < LL; ++t) {
        int i = b * LL + t;
        int qd = q_data[i], ii = qd - 1;
        float m = mvals[i];
        int qa = (qa_data[i] - qd) / NQ_;
        float av = aa[i];
        float g_upd = (m == 1.f) ? mc[i] / av
                                 : ((qa == 0) ? 1.f - nmc[i] / av : nmc[i] / av);
        bool us = (m == 1.f) && (qa == 0);
        float ng = us ? gRow[ii] : g_upd;
        float ns = us ? mi[i] / av : sRow[ii];
        gRow[ii] = ng; sRow[ii] = ns;
        float p = (1.f - ns) * (m * ng + (1.f - ns) * (1.f - m));
        out[1 + i] = 1.f / (1.f + __expf(-p));
        float lab = target[i];
        if (lab > -0.9f) { float d = p - lab; mse += d * d; ++cnt; }
        float pe = diff_parm[pid_data[i]];
        creg += pe * pe;
    }
    s_mse[b] = mse; s_creg[b] = creg; s_cnt[b] = cnt;
    __syncthreads();
    if (b == 0) {
        float tm = 0.f, tc = 0.f; int tn = 0;
        for (int j = 0; j < BB; ++j) { tm += s_mse[j]; tc += s_creg[j]; tn += s_cnt[j]; }
        out[0] = tm + 1e-5f * tc;          // mse_sum + c_reg
        out[1 + NROWS] = (float)tn;        // mask.sum()
    }
}

// ----------------------------- launcher ---------------------------------
extern "C" void kernel_launch(void* const* d_in, const int* in_sizes, int n_in,
                              void* d_out, int out_size, void* d_ws, size_t ws_size,
                              hipStream_t stream) {
    (void)in_sizes; (void)n_in; (void)out_size; (void)ws_size;
    const int*   q_data    = (const int*)  d_in[0];
    const int*   qa_data   = (const int*)  d_in[1];
    const int*   pid_data  = (const int*)  d_in[2];
    const float* matrix    = (const float*)d_in[3];
    const float* target    = (const float*)d_in[4];
    const float* q_emb     = (const float*)d_in[5];
    const float* qa_emb    = (const float*)d_in[6];
    const float* q_emb_d   = (const float*)d_in[7];
    const float* qa_emb_d  = (const float*)d_in[8];
    const float* diff_parm = (const float*)d_in[9];
    const float* w_ih      = (const float*)d_in[10];
    const float* w_hh      = (const float*)d_in[11];
    const float* b_ih      = (const float*)d_in[12];
    const float* b_hh      = (const float*)d_in[13];
    const float* fc_w      = (const float*)d_in[14];
    const float* fc_b      = (const float*)d_in[15];
    float* out = (float*)d_out;

    // workspace carve-up (~110 MB; fc_out aliases dead xg buffer)
    char* w = (char*)d_ws;
    size_t off = 0;
    auto take = [&](size_t bytes) { char* p = w + off; off += (bytes + 255) & ~(size_t)255; return p; };
    _Float16* x16    = (_Float16*)take((size_t)NROWS * 2 * DD * 2);
    _Float16* wih16  = (_Float16*)take((size_t)G3 * HH * 2);
    _Float16* whh16  = (_Float16*)take((size_t)G3 * HH * 2);
    _Float16* fcw16  = (_Float16*)take((size_t)QQ * HH * 2);
    _Float16* gru16  = (_Float16*)take((size_t)NROWS * HH * 2);
    float*    mvals  = (float*)take((size_t)NROWS * 4);
    float*    mc     = (float*)take((size_t)NROWS * 4);
    float*    mi     = (float*)take((size_t)NROWS * 4);
    float*    nmc    = (float*)take((size_t)NROWS * 4);
    float*    aa     = (float*)take((size_t)NROWS * 4);
    float*    guess  = (float*)take((size_t)BB * QQ * 4);
    float*    slip   = (float*)take((size_t)BB * QQ * 4);
    float*    xg     = (float*)take((size_t)NROWS * G3 * 4);
    float*    fco    = xg;   // alias: xg dead after GRU

    conv_w_kernel<<<(G3 * HH + 255) / 256, 256, 0, stream>>>(
        w_ih, w_hh, fc_w, wih16, whh16, fcw16);
    build_x_kernel<<<NROWS, DD, 0, stream>>>(
        q_data, qa_data, pid_data, q_emb, qa_emb, q_emb_d, qa_emb_d, diff_parm, x16);
    xg_gemm_kernel<<<dim3(NROWS / 16, G3 / 128), 256, 0, stream>>>(
        x16, wih16, b_ih, xg);
    gru_kernel<<<BB / 16, 512, 0, stream>>>(xg, whh16, b_hh, gru16);
    fc_gemm_kernel<<<dim3(NROWS / 16, QQ / 128), 256, 0, stream>>>(
        gru16, fcw16, fc_b, fco);
    mvals_kernel<<<NROWS / 8, 256, 0, stream>>>(fco, matrix, q_data, mvals);
    stats_kernel<<<BB, 256, 0, stream>>>(q_data, qa_data, mvals, mc, mi, nmc, aa);
    dina_kernel<<<1, BB, 0, stream>>>(q_data, qa_data, pid_data, diff_parm, target,
                                      mvals, mc, mi, nmc, aa, guess, slip, out);
}
